// DRNN_88527865905382
// MI455X (gfx1250) — matrix operand order repaired and checked by reading, not compiled, not run on hardware
//
#include <hip/hip_runtime.h>

typedef __attribute__((ext_vector_type(16))) _Float16 v16h;
typedef __attribute__((ext_vector_type(8)))  _Float16 v8h;
typedef __attribute__((ext_vector_type(8)))  float    v8f;

#define NWG   16
#define NTHR  128
#define T_LEN 2048

// ---------------- workspace layout (bytes) ----------------
// [0 .. 255]        : barrier cnt (u32), gen (u32)
// [256 .. 16383]    : Hstate, 3584 floats: L0@0 (2x128), L1@256 (3x128),
//                     L2@640 (5x128), L3@1280 (9x256)
// [16384 .. ]       : packed f16 A-fragments, 851968 halves:
//                     L0@0, L1@196608, L2@327680, L3@458752

__device__ __forceinline__ float sigm(float x) { return 1.0f / (1.0f + __expf(-x)); }

// ---------------- weight packing into WMMA A-fragment layout ----------------
// Packed rows per WG: [i-rows S | f-rows S | g-rows S | o-rows S], S = DOUT/16,
// where WG w owns hidden indices [w*S, (w+1)*S).
// A 16x32 f16 fragment layout (ISA 7.12.2): lane L holds M=L%16;
// halves h: K = h + (h>=8 ? 8 : 0) + (L>=16 ? 8 : 0).
template<int DIN, int DOUT>
__global__ void pack_layer(const float* __restrict__ Wih,
                           const float* __restrict__ Whh,
                           _Float16* __restrict__ dst)
{
    constexpr int S    = DOUT / 16;
    constexpr int ROWS = 4 * S;
    constexpr int K    = DIN + DOUT;   // always a multiple of 32 here
    constexpr int KT   = K / 32;
    const long long total = (long long)NWG * ROWS * K;
    long long e = (long long)blockIdx.x * blockDim.x + threadIdx.x;
    if (e >= total) return;
    const int perwg  = ROWS * K;
    const int wg     = (int)(e / perwg);
    const int rem    = (int)(e % perwg);
    const int frag   = rem / 512;
    const int within = rem % 512;
    const int lane   = within / 16;
    const int h      = within % 16;
    const int rt     = frag / KT;
    const int kt     = frag % KT;
    const int mp     = rt * 16 + (lane & 15);      // packed row
    const int hi     = lane >> 4;
    const int kk     = kt * 32 + h + ((h >= 8) ? 8 : 0) + (hi ? 8 : 0);
    const int g      = mp / S, j = mp % S;
    const int orow   = g * DOUT + wg * S + j;      // original matrix row
    const float v = (kk < DIN) ? Wih[(long long)orow * DIN + kk]
                               : Whh[(long long)orow * DOUT + (kk - DIN)];
    dst[e] = (_Float16)v;
}

// ---------------- per-call state init (deterministic under graph replay) ----
__global__ void init_state(float* __restrict__ Hbase, unsigned* __restrict__ bar,
                           const float* __restrict__ h0_0, const float* __restrict__ h0_1,
                           const float* __restrict__ h0_2, const float* __restrict__ h0_3)
{
    const int tid = blockIdx.x * blockDim.x + threadIdx.x;
    if (tid == 0) { bar[0] = 0u; bar[1] = 0u; }
    const int  HOFF[4]  = {0, 256, 640, 1280};
    const int  DOUTa[4] = {128, 128, 128, 256};
    const int  DILa[4]  = {1, 2, 4, 8};
    const float* h0[4]  = {h0_0, h0_1, h0_2, h0_3};
    // total = 1*128 + 2*128 + 4*128 + 8*256 = 2944
    for (int idx = tid; idx < 2944; idx += gridDim.x * blockDim.x) {
        int r = idx, l;
        if      (r < 128)  { l = 0; }
        else if (r < 384)  { l = 1; r -= 128; }
        else if (r < 896)  { l = 2; r -= 384; }
        else               { l = 3; r -= 896; }
        const int dout = DOUTa[l];
        const int k = r / dout, j = r % dout;
        const int P = DILa[l] + 1;
        // h0[k] must be read at step t=k from slot (t+1)%P -> slot (k+1)%P
        Hbase[HOFF[l] + ((k + 1) % P) * dout + j] = h0[l][(long long)k * dout + j];
    }
}

// ---------------- grid-wide sense barrier (persistent 16 WGs) ---------------
__device__ __forceinline__ void gridbar(unsigned* cnt, unsigned* gen)
{
    __threadfence();
    __syncthreads();
    if (threadIdx.x == 0) {
        unsigned g = __hip_atomic_load(gen, __ATOMIC_ACQUIRE, __HIP_MEMORY_SCOPE_AGENT);
        unsigned a = __hip_atomic_fetch_add(cnt, 1u, __ATOMIC_ACQ_REL, __HIP_MEMORY_SCOPE_AGENT);
        if (a == NWG - 1) {
            __hip_atomic_store(cnt, 0u, __ATOMIC_RELAXED, __HIP_MEMORY_SCOPE_AGENT);
            __hip_atomic_fetch_add(gen, 1u, __ATOMIC_ACQ_REL, __HIP_MEMORY_SCOPE_AGENT);
        } else {
            while (__hip_atomic_load(gen, __ATOMIC_ACQUIRE, __HIP_MEMORY_SCOPE_AGENT) == g)
                __builtin_amdgcn_s_sleep(1);
        }
    }
    __syncthreads();
    __threadfence();
}

// ---------------- one LSTM layer step (per workgroup slice) -----------------
template<int DIN, int DOUT, int DILV, int L>
__device__ __forceinline__ void layer_step(
    int t, int wg,
    const _Float16* __restrict__ Apack,  // packed A frags, this layer, all WGs
    const float*    __restrict__ inv,    // layer input vector [DIN] (global)
    float*          __restrict__ Hl,     // h-state for this layer (global, P slots)
    float*          __restrict__ cstL,   // c-state slice (LDS, DILV*S)
    const float*    __restrict__ zbL,    // bias slice (LDS, 4*S)
    _Float16* vbf, float* zbuf,          // LDS scratch
    float* __restrict__ outrow)          // L==3: d_out + t*256
{
    constexpr int S    = DOUT / 16;
    constexpr int ROWS = 4 * S;
    constexpr int K    = DIN + DOUT;
    constexpr int KT   = K / 32;
    constexpr int RT   = ROWS / 16;
    constexpr int P    = DILV + 1;
    const int tid = threadIdx.x;
    const int Ws  = t % P;
    const int Rs  = (t + 1) % P;
    const float* hprev = Hl + Rs * DOUT;

    // stage [x ; h_prev] into LDS as f16
    for (int j = tid; j < K; j += NTHR)
        vbf[j] = (_Float16)((j < DIN) ? inv[j] : hprev[j - DIN]);
    __syncthreads();

    const int wave = tid >> 5;
    const int lane = tid & 31;
    const int hi   = lane >> 4;
    if (wave < RT) {  // wave handles one 16-row tile, full K sweep
        const _Float16* Abase = Apack + (long long)wg * ROWS * K
                              + (long long)wave * KT * 512 + lane * 16;
        v8f acc = {};
#pragma unroll
        for (int kt = 0; kt < KT; ++kt) {
            v16h a = *(const v16h*)(Abase + kt * 512);
            // B 32x16 layout (matches A): halves 0-7 <-> K 8*hi+0..7,
            // halves 8-15 <-> K 16+8*hi+0..7. All N columns identical = v.
            const _Float16* vb = vbf + kt * 32 + (hi ? 8 : 0);
            v8h b0 = *(const v8h*)vb;
            v8h b1 = *(const v8h*)(vb + 16);
            v16h b;
#pragma unroll
            for (int i = 0; i < 8; ++i) { b[i] = b0[i]; b[i + 8] = b1[i]; }
            acc = __builtin_amdgcn_wmma_f32_16x16x32_f16(
                      false, a, false, b, (short)0, acc, false, false);
        }
        // D: VGPR r -> M=r (lane 0), M=r+8 (lane 16); all N equal, take N=0
        if ((lane & 15) == 0) {
            const int mofs = wave * 16 + hi * 8;
#pragma unroll
            for (int r = 0; r < 8; ++r) zbuf[mofs + r] = acc[r];
        }
    }
    __syncthreads();

    // gate nonlinearity for our S hidden units
    if (tid < S) {
        const float zi = zbuf[tid]         + zbL[tid];
        const float zf = zbuf[S + tid]     + zbL[S + tid];
        const float zg = zbuf[2 * S + tid] + zbL[2 * S + tid];
        const float zo = zbuf[3 * S + tid] + zbL[3 * S + tid];
        const float ig = sigm(zi), fg = sigm(zf);
        const float gg = tanhf(zg), og = sigm(zo);
        const int slot = t % DILV;
        const float c  = cstL[slot * S + tid];
        const float c2 = fg * c + ig * gg;
        cstL[slot * S + tid] = c2;
        const float h2 = og * tanhf(c2);
        Hl[Ws * DOUT + wg * S + tid] = h2;
        if (L == 3) outrow[wg * S + tid] = h2;
    }
}

// ---------------- persistent main kernel ------------------------------------
__global__ __launch_bounds__(NTHR)
void drnn_main(const float* __restrict__ x,
               const float* __restrict__ bih0, const float* __restrict__ bhh0,
               const float* __restrict__ bih1, const float* __restrict__ bhh1,
               const float* __restrict__ bih2, const float* __restrict__ bhh2,
               const float* __restrict__ bih3, const float* __restrict__ bhh3,
               const float* __restrict__ c0_0, const float* __restrict__ c0_1,
               const float* __restrict__ c0_2, const float* __restrict__ c0_3,
               const _Float16* __restrict__ packA,
               float* __restrict__ Hbase,
               unsigned* __restrict__ bar,
               float* __restrict__ out)
{
    __shared__ __align__(16) _Float16 vbf[384];
    __shared__ float zbuf[64];
    __shared__ float zb[4][64];
    __shared__ float cst[184];   // L0@0 (1x8), L1@8 (2x8), L2@24 (4x8), L3@56 (8x16)
    const int wg  = blockIdx.x;
    const int tid = threadIdx.x;

    {   // biases (bih+bhh) and initial c-state for our hidden slice
        const float* bi[4] = {bih0, bih1, bih2, bih3};
        const float* bh[4] = {bhh0, bhh1, bhh2, bhh3};
        const float* c0[4] = {c0_0, c0_1, c0_2, c0_3};
        const int Sa[4] = {8, 8, 8, 16}, Da[4] = {128, 128, 128, 256};
        const int Ca[4] = {0, 8, 24, 56}, Dl[4] = {1, 2, 4, 8};
        for (int l = 0; l < 4; ++l) {
            const int S = Sa[l], D = Da[l];
            for (int p = tid; p < 4 * S; p += NTHR) {
                const int g = p / S, j = p % S;
                const int r = g * D + wg * S + j;
                zb[l][p] = bi[l][r] + bh[l][r];
            }
            for (int p = tid; p < Dl[l] * S; p += NTHR) {
                const int k = p / S, j = p % S;
                cst[Ca[l] + p] = c0[l][(long long)k * D + wg * S + j];
            }
        }
    }
    __syncthreads();

    const _Float16* A0 = packA;
    const _Float16* A1 = packA + 196608;
    const _Float16* A2 = packA + 327680;
    const _Float16* A3 = packA + 458752;
    float* H0 = Hbase;        float* H1 = Hbase + 256;
    float* H2 = Hbase + 640;  float* H3 = Hbase + 1280;

    for (int t = 0; t < T_LEN; ++t) {
        layer_step<256, 128, 1, 0>(t, wg, A0, x + (long long)t * 256, H0,
                                   cst + 0,  zb[0], vbf, zbuf, nullptr);
        gridbar(bar, bar + 1);
        layer_step<128, 128, 2, 1>(t, wg, A1, H0 + (t % 2) * 128, H1,
                                   cst + 8,  zb[1], vbf, zbuf, nullptr);
        gridbar(bar, bar + 1);
        layer_step<128, 128, 4, 2>(t, wg, A2, H1 + (t % 3) * 128, H2,
                                   cst + 24, zb[2], vbf, zbuf, nullptr);
        gridbar(bar, bar + 1);
        layer_step<128, 256, 8, 3>(t, wg, A3, H2 + (t % 5) * 128, H3,
                                   cst + 56, zb[3], vbf, zbuf,
                                   out + (long long)t * 256);
        gridbar(bar, bar + 1);
    }
}

// ---------------- host launch ----------------------------------------------
extern "C" void kernel_launch(void* const* d_in, const int* in_sizes, int n_in,
                              void* d_out, int out_size, void* d_ws, size_t ws_size,
                              hipStream_t stream)
{
    (void)in_sizes; (void)n_in; (void)out_size; (void)ws_size;
    const float* x    = (const float*)d_in[0];
    const float* Wih[4], *Whh[4], *bih[4], *bhh[4], *h0[4], *c0[4];
    for (int i = 0; i < 4; ++i) {
        const int b = 1 + 6 * i;
        Wih[i] = (const float*)d_in[b + 0];
        Whh[i] = (const float*)d_in[b + 1];
        bih[i] = (const float*)d_in[b + 2];
        bhh[i] = (const float*)d_in[b + 3];
        h0[i]  = (const float*)d_in[b + 4];
        c0[i]  = (const float*)d_in[b + 5];
    }
    float*     out   = (float*)d_out;
    unsigned*  bar   = (unsigned*)d_ws;
    float*     Hbase = (float*)((char*)d_ws + 256);
    _Float16*  packA = (_Float16*)((char*)d_ws + 16384);

    init_state<<<8, 256, 0, stream>>>(Hbase, bar, h0[0], h0[1], h0[2], h0[3]);
    pack_layer<256, 128><<<768,  256, 0, stream>>>(Wih[0], Whh[0], packA);
    pack_layer<128, 128><<<512,  256, 0, stream>>>(Wih[1], Whh[1], packA + 196608);
    pack_layer<128, 128><<<512,  256, 0, stream>>>(Wih[2], Whh[2], packA + 327680);
    pack_layer<128, 256><<<1536, 256, 0, stream>>>(Wih[3], Whh[3], packA + 458752);
    drnn_main<<<NWG, NTHR, 0, stream>>>(x,
        bih[0], bhh[0], bih[1], bhh[1], bih[2], bhh[2], bih[3], bhh[3],
        c0[0], c0[1], c0[2], c0[3], packA, Hbase, bar, out);
}
